// MultiModalTransformerEncoder_9904194585214
// MI455X (gfx1250) — matrix operand order repaired and checked by reading, not compiled, not run on hardware
//
#include <hip/hip_runtime.h>
#include <hip/hip_bf16.h>
#include <math.h>

// Problem constants (match reference)
#define B_   32
#define S_   512
#define D_   1024
#define H_   16
#define HD_  64
#define L_   6
#define DFF_ 4096
#define P_   64

typedef __attribute__((ext_vector_type(16))) __bf16 v16bf;
typedef __attribute__((ext_vector_type(8)))  float  v8f;

// fp32 -> bf16 bits, round-to-nearest-even
static __device__ __forceinline__ unsigned short bf16u(float f) {
  unsigned u = __builtin_bit_cast(unsigned, f);
  return (unsigned short)((u + 0x7FFFu + ((u >> 16) & 1u)) >> 16);
}
// pack two bf16 (K even, K odd) into one dword
static __device__ __forceinline__ unsigned pack2(float lo, float hi) {
  return (unsigned)bf16u(lo) | ((unsigned)bf16u(hi) << 16);
}

struct U32x8 { uint4 lo, hi; };   // 32 bytes == v16bf

// ---------------------------------------------------------------------------
// Staging helpers: convert f32 global tiles -> packed bf16-pair LDS tiles laid
// out in WMMA fragment order (pair p of row r = {elem 2p, elem 2p+1} along K).
// Row stride 20 dwords => ds_load_b128 fragment gathers are bank-conflict-free.
// ---------------------------------------------------------------------------
static __device__ __forceinline__ void stage_A(unsigned (*As)[20], const float* Ab, long lda,
                                               int m0, int k0, int M, int t) {
  const int r  = t >> 3;          // 0..31, 4 passes -> 128 rows
  const int c4 = (t & 7) * 4;     // K offset within 32-slab
#pragma unroll
  for (int rr = r; rr < 128; rr += 32) {
    const int gr = m0 + rr;
    float4 v = {0.f, 0.f, 0.f, 0.f};
    if (gr < M) v = *reinterpret_cast<const float4*>(Ab + (long)gr * lda + k0 + c4);
    As[rr][(c4 >> 1) + 0] = pack2(v.x, v.y);
    As[rr][(c4 >> 1) + 1] = pack2(v.z, v.w);
  }
}

// B is K x N row major: read two adjacent K rows, pack vertically per column
static __device__ __forceinline__ void stage_Bn(unsigned (*Bs)[20], const float* Bb, long ldb,
                                                int n0, int k0, int N, int t) {
  const int p  = t & 15;          // K pair 0..15
  const int n4 = (t >> 4) * 4;    // 0..60, 2 passes -> 128 cols
#pragma unroll
  for (int nn = n4; nn < 128; nn += 64) {
    const int gc = n0 + nn;
    float4 r0 = {0.f, 0.f, 0.f, 0.f}, r1 = r0;
    if (gc < N) {
      const float* bq = Bb + (long)(k0 + 2 * p) * ldb + gc;
      r0 = *reinterpret_cast<const float4*>(bq);
      r1 = *reinterpret_cast<const float4*>(bq + ldb);
    }
    Bs[nn + 0][p] = pack2(r0.x, r1.x);
    Bs[nn + 1][p] = pack2(r0.y, r1.y);
    Bs[nn + 2][p] = pack2(r0.z, r1.z);
    Bs[nn + 3][p] = pack2(r0.w, r1.w);
  }
}

// B^T is N x K row major: read along K (coalesced), pack pairs directly
static __device__ __forceinline__ void stage_Bt(unsigned (*Bs)[20], const float* Bt, long ldb,
                                                int n0, int k0, int N, int t) {
  const int nr = t >> 3;          // 0..31, 4 passes -> 128 cols
  const int k4 = (t & 7) * 4;
#pragma unroll
  for (int nn = nr; nn < 128; nn += 32) {
    float4 v = {0.f, 0.f, 0.f, 0.f};
    if (n0 + nn < N) v = *reinterpret_cast<const float4*>(Bt + (long)(n0 + nn) * ldb + k0 + k4);
    Bs[nn][(k4 >> 1) + 0] = pack2(v.x, v.y);
    Bs[nn][(k4 >> 1) + 1] = pack2(v.z, v.w);
  }
}

// ---------------------------------------------------------------------------
// LDS-staged, double-buffered batched WMMA GEMM:
//     C[z] = act( A[z] @ B[z] + bias + radd + addC )
// Block: 256 threads (8 wave32s), tile 128(M) x 128(N) x 32(K), ping-pong LDS.
// Each wave owns a 32x64 strip: 2 A-fragments x 4 B-fragments -> 8 WMMAs per
// K-step (v_wmma_f32_16x16x32_bf16, f32 accumulate), 12 ds_load_b128 gathers.
// Batch offset of tensor T = (z/innerB)*sT1 + (z%innerB)*sT2.
// Fragment gather (CDNA5 ISA 7.12.2, wave32, lane m/n = lane&15, half=lane>>4):
//   A vgpr v -> pair (v>>2)*8 + half*4 + (v&3)  => 2x ds_load_b128 / fragment
//   B vgpr v -> pair (v>>2)*8 + half*4 + (v&3)  => 2x ds_load_b128 / fragment
// ---------------------------------------------------------------------------
template <int ACT, bool TRANSB>
__global__ void __launch_bounds__(256)
wmma_gemm_kernel(const float* __restrict__ A, long lda, long sA1, long sA2,
                 const float* __restrict__ Bm, long ldb, long sB1, long sB2,
                 float* __restrict__ C, long ldc, long sC1, long sC2,
                 const float* __restrict__ bias,           // per-col or null
                 const float* __restrict__ radd, int rgrp, // broadcast add per row-group
                 const float* __restrict__ addC,           // elementwise add (C layout)
                 int M, int N, int K, int innerB) {
  __shared__ unsigned As[2][128][20];   // 20 KB (ping-pong)
  __shared__ unsigned Bs[2][128][20];   // 20 KB

  const int t    = threadIdx.x;
  const int lane = t & 31;
  const int w    = t >> 5;
  const int l16  = lane & 15;
  const int half = lane >> 4;

  const int m0 = blockIdx.y * 128;
  const int n0 = blockIdx.x * 128;

  const int z  = blockIdx.z;
  const int zo = z / innerB, zi = z % innerB;
  const float* Ab = A  + (long)zo * sA1 + (long)zi * sA2;
  const float* Bb = Bm + (long)zo * sB1 + (long)zi * sB2;
  float*       Cb = C  + (long)zo * sC1 + (long)zi * sC2;
  const float* Db = addC ? addC + (long)zo * sC1 + (long)zi * sC2 : nullptr;

  const int arow0 = (w & 3) * 32 + l16;   // A fragment rows: +0, +16
  const int bcol0 = (w >> 2) * 64 + l16;  // B fragment cols: +0,16,32,48

  v8f acc[2][4];
#pragma unroll
  for (int mi = 0; mi < 2; ++mi)
#pragma unroll
    for (int nf = 0; nf < 4; ++nf)
#pragma unroll
      for (int i = 0; i < 8; ++i) acc[mi][nf][i] = 0.f;

  // prologue: stage slab 0 into buffer 0
  stage_A(As[0], Ab, lda, m0, 0, M, t);
  if (TRANSB) stage_Bt(Bs[0], Bb, ldb, n0, 0, N, t);
  else        stage_Bn(Bs[0], Bb, ldb, n0, 0, N, t);
  __syncthreads();

  for (int k0 = 0; k0 < K; k0 += 32) {
    const int buf = (k0 >> 5) & 1;
    // stage next slab into the idle buffer (global loads issue ahead of WMMAs)
    if (k0 + 32 < K) {
      stage_A(As[buf ^ 1], Ab, lda, m0, k0 + 32, M, t);
      if (TRANSB) stage_Bt(Bs[buf ^ 1], Bb, ldb, n0, k0 + 32, N, t);
      else        stage_Bn(Bs[buf ^ 1], Bb, ldb, n0, k0 + 32, N, t);
      // speculative prefetch of the slab after next (global_prefetch_b8)
      if (k0 + 64 < K && m0 + (t >> 3) < M)
        __builtin_prefetch(Ab + (long)(m0 + (t >> 3)) * lda + k0 + 64 + (t & 7) * 4, 0, 0);
    }

    // ---- compute from current buffer: 12x ds_load_b128, 8x WMMA ----
    const unsigned (*Ac)[20] = As[buf];
    const unsigned (*Bc)[20] = Bs[buf];

    U32x8 ua;
    ua.lo = *reinterpret_cast<const uint4*>(&Ac[arow0][half * 4]);
    ua.hi = *reinterpret_cast<const uint4*>(&Ac[arow0][8 + half * 4]);
    const v16bf a0 = __builtin_bit_cast(v16bf, ua);
    ua.lo = *reinterpret_cast<const uint4*>(&Ac[arow0 + 16][half * 4]);
    ua.hi = *reinterpret_cast<const uint4*>(&Ac[arow0 + 16][8 + half * 4]);
    const v16bf a1 = __builtin_bit_cast(v16bf, ua);

#pragma unroll
    for (int nf = 0; nf < 4; ++nf) {
      U32x8 ub;
      ub.lo = *reinterpret_cast<const uint4*>(&Bc[bcol0 + nf * 16][half * 4]);
      ub.hi = *reinterpret_cast<const uint4*>(&Bc[bcol0 + nf * 16][8 + half * 4]);
      const v16bf bfv = __builtin_bit_cast(v16bf, ub);
      acc[0][nf] = __builtin_amdgcn_wmma_f32_16x16x32_bf16(false, a0, false, bfv,
                                                           (short)0, acc[0][nf], false, false);
      acc[1][nf] = __builtin_amdgcn_wmma_f32_16x16x32_bf16(false, a1, false, bfv,
                                                           (short)0, acc[1][nf], false, false);
    }
    __syncthreads();   // next buffer fully staged + current fully consumed
  }

  // ---- epilogue ----
#pragma unroll
  for (int mi = 0; mi < 2; ++mi) {
#pragma unroll
    for (int nf = 0; nf < 4; ++nf) {
#pragma unroll
      for (int i = 0; i < 8; ++i) {
        const int grow = m0 + (w & 3) * 32 + mi * 16 + i + half * 8;
        const int gcol = n0 + (w >> 2) * 64 + nf * 16 + l16;
        if (grow >= M || gcol >= N) continue;
        float v = acc[mi][nf][i];
        if (bias) v += bias[gcol];
        if (radd) v += radd[(long)(grow / rgrp) * N + gcol];
        if (addC) v += Db[(long)grow * ldc + gcol];
        if (ACT == 1)      v = fmaxf(v, 0.f);
        else if (ACT == 2) v = 1.f / (1.f + __expf(-v));
        else if (ACT == 3) v = tanhf(v);
        Cb[(long)grow * ldc + gcol] = v;
      }
    }
  }
}

template <int ACT, bool TRANSB>
static inline void gemm(hipStream_t st,
                        const float* A, long lda, long sA1, long sA2,
                        const float* Bm, long ldb, long sB1, long sB2,
                        float* C, long ldc, long sC1, long sC2,
                        const float* bias, const float* radd, int rgrp,
                        const float* addC, int M, int N, int K, int nb, int innerB) {
  dim3 g((N + 127) / 128, (M + 127) / 128, nb), blk(256);
  wmma_gemm_kernel<ACT, TRANSB><<<g, blk, 0, st>>>(A, lda, sA1, sA2, Bm, ldb, sB1, sB2,
                                                   C, ldc, sC1, sC2, bias, radd, rgrp,
                                                   addC, M, N, K, innerB);
}

// ---------------------------------------------------------------------------
// Embedding * sqrt(D) + sinusoidal positional encoding. One block per token.
// ---------------------------------------------------------------------------
__global__ void embed_pe_kernel(const int* __restrict__ tok,
                                const float* __restrict__ emb,
                                float* __restrict__ x) {
  const long row = blockIdx.x;             // [0, B*S)
  const int  s   = (int)(row % S_);
  const int  tk  = tok[row];
  const float p  = (float)s;
  const int  t   = threadIdx.x;
#pragma unroll
  for (int j = 0; j < 4; ++j) {
    const int d = t * 4 + j;
    const float div = __expf(-(float)((d >> 1) << 1) * (9.2103403719761836f / (float)D_));
    const float pe  = (d & 1) ? __cosf(p * div) : __sinf(p * div);
    x[row * D_ + d] = emb[(long)tk * D_ + d] * 32.0f + pe;  // sqrt(1024)=32
  }
}

// ---------------------------------------------------------------------------
// Row softmax over S=512 with 1/sqrt(HD) scale + key-mask bias. In place.
// ---------------------------------------------------------------------------
__global__ void softmax_mask_kernel(float* __restrict__ sc, const int* __restrict__ mask) {
  const long row = blockIdx.x;             // [0, B*H*S)
  const int  b   = (int)(row / ((long)H_ * S_));
  float* r = sc + row * S_;
  const int t = threadIdx.x;
  __shared__ float red[256];

  float v0 = r[t]       * 0.125f + (mask[b * S_ + t]       ? 0.f : -1e9f);
  float v1 = r[t + 256] * 0.125f + (mask[b * S_ + t + 256] ? 0.f : -1e9f);

  red[t] = fmaxf(v0, v1);
  __syncthreads();
  for (int o = 128; o > 0; o >>= 1) { if (t < o) red[t] = fmaxf(red[t], red[t + o]); __syncthreads(); }
  const float m = red[0];
  __syncthreads();

  const float e0 = __expf(v0 - m), e1 = __expf(v1 - m);
  red[t] = e0 + e1;
  __syncthreads();
  for (int o = 128; o > 0; o >>= 1) { if (t < o) red[t] += red[t + o]; __syncthreads(); }
  const float inv = 1.f / red[0];
  r[t] = e0 * inv;
  r[t + 256] = e1 * inv;
}

// ---------------------------------------------------------------------------
// x = LayerNorm(x + y) * g + b, in place on x. One block per row of D=1024.
// ---------------------------------------------------------------------------
__global__ void add_ln_kernel(float* __restrict__ x, const float* __restrict__ y,
                              const float* __restrict__ g, const float* __restrict__ bb) {
  const long row = blockIdx.x;
  const int  t   = threadIdx.x;
  float* xr = x + row * D_;
  const float* yr = y + row * D_;
  __shared__ float red[256];

  float v[4]; float s = 0.f;
#pragma unroll
  for (int j = 0; j < 4; ++j) { const int d = t * 4 + j; v[j] = xr[d] + yr[d]; s += v[j]; }
  red[t] = s;
  __syncthreads();
  for (int o = 128; o > 0; o >>= 1) { if (t < o) red[t] += red[t + o]; __syncthreads(); }
  const float mu = red[0] / (float)D_;
  __syncthreads();

  float q = 0.f;
#pragma unroll
  for (int j = 0; j < 4; ++j) { const float dlt = v[j] - mu; q += dlt * dlt; }
  red[t] = q;
  __syncthreads();
  for (int o = 128; o > 0; o >>= 1) { if (t < o) red[t] += red[t + o]; __syncthreads(); }
  const float inv = rsqrtf(red[0] / (float)D_ + 1e-5f);
#pragma unroll
  for (int j = 0; j < 4; ++j) {
    const int d = t * 4 + j;
    xr[d] = (v[j] - mu) * inv * g[d] + bb[d];
  }
}

// att[b,s,d] *= gate[b,d]
__global__ void rowgroup_mul_kernel(float* __restrict__ a, const float* __restrict__ gate) {
  const long idx = (long)blockIdx.x * 256 + threadIdx.x;   // [0, B*S*D)
  const int  d   = (int)(idx % D_);
  const int  b   = (int)((idx / D_) / S_);
  a[idx] *= gate[(long)b * D_ + d];
}

// pooled[b,d] = sum_s x*m / clamp(sum m)
__global__ void pool_mask_kernel(const float* __restrict__ x, const int* __restrict__ mask,
                                 float* __restrict__ pooled) {
  const int idx = blockIdx.x * 256 + threadIdx.x;          // [0, B*D)
  const int b = idx / D_, d = idx % D_;
  float s = 0.f, ms = 0.f;
  for (int t = 0; t < S_; ++t) {
    const float m = (float)mask[b * S_ + t];
    s  += x[((long)b * S_ + t) * D_ + d] * m;
    ms += m;
  }
  pooled[idx] = s / fmaxf(ms, 1e-9f);
}

// ---------------------------------------------------------------------------
extern "C" void kernel_launch(void* const* d_in, const int* in_sizes, int n_in,
                              void* d_out, int out_size, void* d_ws, size_t ws_size,
                              hipStream_t stream) {
  (void)in_sizes; (void)n_in; (void)out_size; (void)ws_size;

  const int*   tok   = (const int*)d_in[0];
  const int*   amask = (const int*)d_in[1];
  const float* pv    = (const float*)d_in[2];
  const float* emb   = (const float*)d_in[3];
  const float* Wq0 = (const float*)d_in[4],  *bq0 = (const float*)d_in[5];
  const float* Wk0 = (const float*)d_in[6],  *bk0 = (const float*)d_in[7];
  const float* Wv0 = (const float*)d_in[8],  *bv0 = (const float*)d_in[9];
  const float* Wo0 = (const float*)d_in[10], *bo0 = (const float*)d_in[11];
  const float* Wpq0= (const float*)d_in[12], *bpq0= (const float*)d_in[13];
  const float* Wpk0= (const float*)d_in[14], *bpk0= (const float*)d_in[15];
  const float* Wpg0= (const float*)d_in[16], *bpg0= (const float*)d_in[17];
  const float* Wf10= (const float*)d_in[18], *bf10= (const float*)d_in[19];
  const float* Wf20= (const float*)d_in[20], *bf20= (const float*)d_in[21];
  const float* g10 = (const float*)d_in[22], *b10 = (const float*)d_in[23];
  const float* g20 = (const float*)d_in[24], *b20 = (const float*)d_in[25];
  const float* Wp1 = (const float*)d_in[26], *bp1 = (const float*)d_in[27];
  const float* Wp2 = (const float*)d_in[28], *bp2 = (const float*)d_in[29];
  const float* Wo1 = (const float*)d_in[30], *bo1 = (const float*)d_in[31];
  const float* Wo2 = (const float*)d_in[32], *bo2 = (const float*)d_in[33];

  // ---- workspace layout (floats) ----
  float* ws = (float*)d_ws;
  const long BSD = (long)B_ * S_ * D_;                 // 16.8M
  float* X   = ws;
  float* Q   = X   + BSD;
  float* Kt  = Q   + BSD;
  float* Vt  = Kt  + BSD;
  float* ATT = Vt  + BSD;
  float* FFH = ATT + BSD;                              // B*S*DFF
  float* SC  = FFH + (long)B_ * S_ * DFF_;             // B*H*S*S
  float* pq  = SC  + (long)B_ * H_ * S_ * S_;
  float* pk  = pq  + (long)B_ * D_;
  float* gate= pk  + (long)B_ * D_;
  float* pool= gate+ (long)B_ * D_;
  float* t1  = pool+ (long)B_ * D_;
  float* t2  = t1  + (long)B_ * D_;

  // ---- embedding + positional encoding ----
  embed_pe_kernel<<<dim3(B_ * S_), dim3(256), 0, stream>>>(tok, emb, X);

  const long SD  = (long)S_ * D_;
  const long HSS = (long)H_ * S_ * S_;
  const long SS  = (long)S_ * S_;

  for (int l = 0; l < L_; ++l) {
    const float* Wq_ = Wq0 + (long)l * D_ * D_;   const float* bq_ = bq0 + (long)l * D_;
    const float* Wk_ = Wk0 + (long)l * D_ * D_;   const float* bk_ = bk0 + (long)l * D_;
    const float* Wv_ = Wv0 + (long)l * D_ * D_;   const float* bv_ = bv0 + (long)l * D_;
    const float* Wo_ = Wo0 + (long)l * D_ * D_;   const float* bo_ = bo0 + (long)l * D_;
    const float* Wpq_= Wpq0+ (long)l * P_ * D_;   const float* bpq_= bpq0+ (long)l * D_;
    const float* Wpk_= Wpk0+ (long)l * P_ * D_;   const float* bpk_= bpk0+ (long)l * D_;
    const float* Wpg_= Wpg0+ (long)l * P_ * D_;   const float* bpg_= bpg0+ (long)l * D_;
    const float* Wf1_= Wf10+ (long)l * D_ * DFF_; const float* bf1_= bf10+ (long)l * DFF_;
    const float* Wf2_= Wf20+ (long)l * DFF_ * D_; const float* bf2_= bf20+ (long)l * D_;
    const float* g1_ = g10 + (long)l * D_;        const float* b1_ = b10 + (long)l * D_;
    const float* g2_ = g20 + (long)l * D_;        const float* b2_ = b20 + (long)l * D_;

    // property projections: pq/pk plain, gate = sigmoid(...)
    gemm<0,false>(stream, pv,P_,0,0, Wpq_,D_,0,0, pq,D_,0,0, bpq_,nullptr,1,nullptr, B_,D_,P_,1,1);
    gemm<0,false>(stream, pv,P_,0,0, Wpk_,D_,0,0, pk,D_,0,0, bpk_,nullptr,1,nullptr, B_,D_,P_,1,1);
    gemm<2,false>(stream, pv,P_,0,0, Wpg_,D_,0,0, gate,D_,0,0, bpg_,nullptr,1,nullptr, B_,D_,P_,1,1);

    // Q/K/V projections (Q,K get the broadcast property add fused: rgrp = S)
    gemm<0,false>(stream, X,D_,0,0, Wq_,D_,0,0, Q ,D_,0,0, bq_, pq, S_, nullptr, B_*S_,D_,D_,1,1);
    gemm<0,false>(stream, X,D_,0,0, Wk_,D_,0,0, Kt,D_,0,0, bk_, pk, S_, nullptr, B_*S_,D_,D_,1,1);
    gemm<0,false>(stream, X,D_,0,0, Wv_,D_,0,0, Vt,D_,0,0, bv_, nullptr,1,nullptr, B_*S_,D_,D_,1,1);

    // scores[b,h] = Q[b,:,h,:] @ K[b,:,h,:]^T   (batched, B^T layout)
    gemm<0,true >(stream, Q,D_,SD,HD_, Kt,D_,SD,HD_, SC,S_,HSS,SS,
                  nullptr,nullptr,1,nullptr, S_,S_,HD_, B_*H_, H_);
    softmax_mask_kernel<<<dim3(B_ * H_ * S_), dim3(256), 0, stream>>>(SC, amask);

    // att[b,:,h,:] = attn[b,h] @ V[b,:,h,:]
    gemm<0,false>(stream, SC,S_,HSS,SS, Vt,D_,SD,HD_, ATT,D_,SD,HD_,
                  nullptr,nullptr,1,nullptr, S_,HD_,S_, B_*H_, H_);

    // gated output projection + residual LN
    rowgroup_mul_kernel<<<dim3((unsigned)(BSD / 256)), dim3(256), 0, stream>>>(ATT, gate);
    gemm<0,false>(stream, ATT,D_,0,0, Wo_,D_,0,0, Q,D_,0,0, bo_,nullptr,1,nullptr, B_*S_,D_,D_,1,1);
    add_ln_kernel<<<dim3(B_ * S_), dim3(256), 0, stream>>>(X, Q, g1_, b1_);

    // FFN + residual LN
    gemm<1,false>(stream, X,D_,0,0, Wf1_,DFF_,0,0, FFH,DFF_,0,0, bf1_,nullptr,1,nullptr, B_*S_,DFF_,D_,1,1);
    gemm<0,false>(stream, FFH,DFF_,0,0, Wf2_,D_,0,0, Q,D_,0,0, bf2_,nullptr,1,nullptr, B_*S_,D_,DFF_,1,1);
    add_ln_kernel<<<dim3(B_ * S_), dim3(256), 0, stream>>>(X, Q, g2_, b2_);
  }

  // ---- pooling + heads ----
  pool_mask_kernel<<<dim3(B_ * D_ / 256), dim3(256), 0, stream>>>(X, amask, pool);
  gemm<1,false>(stream, pv,P_,0,0, Wp1,D_,0,0, t1,D_,0,0, bp1,nullptr,1,nullptr, B_,D_,P_,1,1);
  gemm<0,false>(stream, t1,D_,0,0, Wp2,D_,0,0, t2,D_,0,0, bp2,nullptr,1, pool,  B_,D_,D_,1,1); // comb
  gemm<3,false>(stream, t2,D_,0,0, Wo1,D_,0,0, t1,D_,0,0, bo1,nullptr,1,nullptr, B_,D_,D_,1,1); // tanh
  gemm<0,false>(stream, t1,D_,0,0, Wo2,D_,0,0, (float*)d_out,D_,0,0, bo2,nullptr,1,nullptr, B_,D_,D_,1,1);
}
